// EAConv_43258910605894
// MI455X (gfx1250) — matrix operand order repaired and checked by reading, not compiled, not run on hardware
//
#include <hip/hip_runtime.h>

// EAConv routing on MI455X (gfx1250, wave32).
// One node per 32-thread workgroup (single-wave -> barriers are hardware NOPs).
// z tile staged in LDS as f16 in two layouts (row-major zsh and transposed
// zshT) so both per-node einsums run on v_wmma_f32_16x16x32_f16 with all
// operand words built from single aligned 32-bit LDS loads + one cndmask.
// Workload is gather/L2-bound (~100MB HBM total), so f16 WMMA inputs with
// f32 accumulation are free; softmax/normalize/residual stay f32.

typedef __attribute__((ext_vector_type(16))) _Float16 v16h;
typedef __attribute__((ext_vector_type(8)))  float    v8f;

#define KFAC   4
#define DD     16
#define DIM    64        // KFAC * DD
#define MNB    16        // neighbors
#define TSTEPS 3
#define AGG    0.5f
#define EPSN   1e-12f

union ABReg { v16h v; _Float16 h[16]; unsigned u[8]; };
union Pk    { _Float16 h[2]; unsigned u; };

__device__ __forceinline__ unsigned ldsu(const _Float16* p) {
    unsigned v; __builtin_memcpy(&v, p, 4); return v;
}
__device__ __forceinline__ void stsu(_Float16* p, unsigned v) {
    __builtin_memcpy(p, &v, 4);
}

__global__ __launch_bounds__(32) void ea_route_kernel(
    const float* __restrict__ x_t,          // [n, 64] slice for this t
    const long long* __restrict__ nbr_t,    // [n, 16] slice for this t
    float* __restrict__ out,                // [T, n, 64] full output
    const int* __restrict__ p_maxiter,
    int t, int n)
{
    const int lane = threadIdx.x;
    const int node = blockIdx.x;            // grid == n
    const int max_iter = p_maxiter[0];

    __shared__ _Float16 zsh [MNB * DIM];    // z[m][k*16+d]        (f16)
    __shared__ _Float16 zshT[KFAC * DD * MNB]; // z[k*16+d][m]     (f16)
    __shared__ float    xsh [DIM];          // xk (f32, normalized own row)
    __shared__ float    ush [DIM];          // u[k][d] (f32)
    __shared__ _Float16 ush16[DIM];         // f16 copy of normalized u
    __shared__ float    lsh [MNB * KFAC];   // logits[m][k]
    __shared__ _Float16 pshT[KFAC * MNB];   // p[k][m] (f16)
    __shared__ int      nsh [MNB];

    // ---- 1. neighbor indices ---------------------------------------------
    if (lane < MNB) {
        long long v = nbr_t[(size_t)node * MNB + lane];
        nsh[lane] = (v >= 0 && v < (long long)n) ? (int)v : n;   // n == pad row
    }
    __syncthreads();

    // ---- 2. gather + per-factor normalize, fully in registers ------------
    // lane covers row mm = lane/2, half hf = lane&1 -> exactly two complete
    // (m,k) factor groups of 16 floats each; no f32 LDS staging needed.
    {
        const int mm = lane >> 1;
        const int hf = lane & 1;
        const int idx  = nsh[mm];
        const bool ok  = (idx < n);
        const int idxc = ok ? idx : 0;
        const float4* src = (const float4*)x_t + (size_t)idxc * 16 + hf * 8;
        float v[32];
        #pragma unroll
        for (int j = 0; j < 8; ++j) {
            float4 q = src[j];
            if (!ok) q = make_float4(0.f, 0.f, 0.f, 0.f);
            v[j * 4 + 0] = q.x; v[j * 4 + 1] = q.y;
            v[j * 4 + 2] = q.z; v[j * 4 + 3] = q.w;
        }
        #pragma unroll
        for (int g = 0; g < 2; ++g) {
            const int k = hf * 2 + g;                 // global factor index
            float ss = 0.f;
            #pragma unroll
            for (int d = 0; d < DD; ++d) ss += v[g * 16 + d] * v[g * 16 + d];
            const float sc = 1.f / fmaxf(sqrtf(ss), EPSN);
            #pragma unroll
            for (int d = 0; d < DD; d += 2) {         // packed row-major store
                Pk pk;
                pk.h[0] = (_Float16)(v[g * 16 + d]     * sc);
                pk.h[1] = (_Float16)(v[g * 16 + d + 1] * sc);
                stsu(&zsh[mm * DIM + k * DD + d], pk.u);
                zshT[(k * DD + d)     * MNB + mm] = pk.h[0];  // transposed copy
                zshT[(k * DD + d + 1) * MNB + mm] = pk.h[1];
            }
        }
    }
    // own row -> xk (f32), one factor group per lane 0..3
    if (lane < KFAC) {
        const float4* src = (const float4*)(x_t + (size_t)node * DIM + lane * DD);
        float v[16]; float ss = 0.f;
        #pragma unroll
        for (int j = 0; j < 4; ++j) {
            float4 q = src[j];
            v[j * 4 + 0] = q.x; v[j * 4 + 1] = q.y;
            v[j * 4 + 2] = q.z; v[j * 4 + 3] = q.w;
        }
        #pragma unroll
        for (int d = 0; d < DD; ++d) ss += v[d] * v[d];
        const float sc = 1.f / fmaxf(sqrtf(ss), EPSN);
        #pragma unroll
        for (int d = 0; d < DD; ++d) xsh[lane * DD + d] = v[d] * sc;
    }
    __syncthreads();

    // ---- 3. iteration 0: uniform p = 1/K ---------------------------------
    #pragma unroll
    for (int s = 0; s < 2; ++s) {
        const int kd = lane * 2 + s;
        float acc = 0.f;
        #pragma unroll
        for (int mw = 0; mw < 8; ++mw) {
            Pk pk; pk.u = ldsu(&zshT[kd * MNB + mw * 2]);
            acc += (float)pk.h[0] + (float)pk.h[1];
        }
        ush[kd] = acc * (1.f / KFAC) + xsh[kd];
    }
    __syncthreads();
    if (max_iter > 1) {
        if (lane < KFAC) {               // normalize u, refresh f16 copy
            float* row = &ush[lane * DD];
            float ss = 0.f;
            #pragma unroll
            for (int d = 0; d < DD; ++d) ss += row[d] * row[d];
            const float sc = 1.f / fmaxf(sqrtf(ss), EPSN);
            #pragma unroll
            for (int d = 0; d < DD; d += 2) {
                const float a = row[d] * sc, b = row[d + 1] * sc;
                row[d] = a; row[d + 1] = b;
                Pk pk; pk.h[0] = (_Float16)a; pk.h[1] = (_Float16)b;
                stsu(&ush16[lane * DD + d], pk.u);
            }
        }
        __syncthreads();
    }

    const int halfsel = lane >> 4;
    const int col     = lane & 15;

    // ---- 4. routing iterations (WMMA path) --------------------------------
    for (int it = 1; it < max_iter; ++it) {
        // 4a. logits[m,k]: A[m, k'*16+d] = z, B[k'*16+d, j] = u (block-diag)
        #pragma unroll
        for (int kb = 0; kb < KFAC; kb += 2) {
            ABReg A, B;
            const int mrow = col;
            unsigned uw[8];
            #pragma unroll
            for (int w = 0; w < 8; ++w) {
                const int Kb = (halfsel << 4) + (w << 1);
                A.u[w]  = ldsu(&zsh[mrow * DIM + kb * DD + Kb]);
                uw[w]   = ldsu(&ush16[(kb + halfsel) * DD + 2 * w]);   // k' = halfsel
            }
            const bool on = (col == halfsel);
            #pragma unroll
            for (int w = 0; w < 8; ++w) B.u[w] = on ? uw[w] : 0u;
            v8f c = {};
            c = __builtin_amdgcn_wmma_f32_16x16x32_f16(
                    false, A.v, false, B.v, (short)0, c, false, false);
            if (col < 2) {
                #pragma unroll
                for (int r = 0; r < 8; ++r)
                    lsh[(r + (halfsel << 3)) * KFAC + kb + col] = c[r];
            }
        }
        __syncthreads();

        // 4b. softmax over K per m -> pshT (f16, transposed)
        if (lane < MNB) {
            const float l0 = lsh[lane * 4 + 0];
            const float l1 = lsh[lane * 4 + 1];
            const float l2 = lsh[lane * 4 + 2];
            const float l3 = lsh[lane * 4 + 3];
            const float mx = fmaxf(fmaxf(l0, l1), fmaxf(l2, l3));
            const float e0 = __expf(l0 - mx), e1 = __expf(l1 - mx);
            const float e2 = __expf(l2 - mx), e3 = __expf(l3 - mx);
            const float inv = 1.f / (e0 + e1 + e2 + e3);
            pshT[0 * MNB + lane] = (_Float16)(e0 * inv);
            pshT[1 * MNB + lane] = (_Float16)(e1 * inv);
            pshT[2 * MNB + lane] = (_Float16)(e2 * inv);
            pshT[3 * MNB + lane] = (_Float16)(e3 * inv);
        }
        __syncthreads();

        // 4c. u[k,d]: A[d, k'*16+m] = z^T, B[k'*16+m, j] = p (block-diag)
        #pragma unroll
        for (int kb = 0; kb < KFAC; kb += 2) {
            ABReg A, B;
            const int drow = col;
            unsigned uw[8];
            #pragma unroll
            for (int w = 0; w < 8; ++w) {
                const int kp = halfsel;
                const int mm = w << 1;
                A.u[w] = ldsu(&zshT[((kb + kp) * DD + drow) * MNB + mm]);
                uw[w]  = ldsu(&pshT[(kb + halfsel) * MNB + 2 * w]);    // k' = halfsel
            }
            const bool on = (col == halfsel);
            #pragma unroll
            for (int w = 0; w < 8; ++w) B.u[w] = on ? uw[w] : 0u;
            v8f c = {};
            c = __builtin_amdgcn_wmma_f32_16x16x32_f16(
                    false, A.v, false, B.v, (short)0, c, false, false);
            if (col < 2) {
                const int k = kb + col;
                #pragma unroll
                for (int r = 0; r < 8; ++r) {
                    const int dr = r + (halfsel << 3);
                    ush[k * DD + dr] = c[r] + xsh[k * DD + dr];   // + xk residual
                }
            }
        }
        __syncthreads();

        // 4d. inter-iteration normalize (+ refresh f16 copy for next logits)
        if (it < max_iter - 1) {
            if (lane < KFAC) {
                float* row = &ush[lane * DD];
                float ss = 0.f;
                #pragma unroll
                for (int d = 0; d < DD; ++d) ss += row[d] * row[d];
                const float sc = 1.f / fmaxf(sqrtf(ss), EPSN);
                #pragma unroll
                for (int d = 0; d < DD; d += 2) {
                    const float a = row[d] * sc, b = row[d + 1] * sc;
                    row[d] = a; row[d + 1] = b;
                    Pk pk; pk.h[0] = (_Float16)a; pk.h[1] = (_Float16)b;
                    stsu(&ush16[lane * DD + d], pk.u);
                }
            }
            __syncthreads();
        }
    }

    // ---- 5. temporal mixing epilogue + store ------------------------------
    #pragma unroll
    for (int s = 0; s < 2; ++s) {
        const int jj = lane * 2 + s;
        const float u = ush[jj];
        float e;
        if (t == 0) {
            e = u;
        } else {
            float acc = 0.f;
            for (int sp = 0; sp < t; ++sp) {
                const float w = 1.f / (1.f + __expf(-(float)sp));   // sigmoid(sp)
                acc += w * out[((size_t)sp * n + node) * DIM + jj];
            }
            e = acc / (float)t * AGG + u * (1.f - AGG);
        }
        out[((size_t)t * n + node) * DIM + jj] = e;
    }
}

extern "C" void kernel_launch(void* const* d_in, const int* in_sizes, int n_in,
                              void* d_out, int out_size, void* d_ws, size_t ws_size,
                              hipStream_t stream) {
    (void)d_ws; (void)ws_size; (void)n_in; (void)in_sizes;
    const float*     x_all   = (const float*)d_in[0];      // [T, 1, n, 64]
    const long long* nbr_all = (const long long*)d_in[1];  // [T, n, 16]
    const int*       p_maxit = (const int*)d_in[2];
    float*           out     = (float*)d_out;              // [T, 1, n, 64]

    const int T = TSTEPS;                     // fixed by reference setup
    const int n = out_size / (T * DIM);

    dim3 block(32);                           // single wave -> barriers are NOPs
    dim3 grid(n);                             // one node per workgroup

    for (int t = 0; t < T; ++t) {             // sequential: t reads t' < t outputs
        ea_route_kernel<<<grid, block, 0, stream>>>(
            x_all   + (size_t)t * n * DIM,
            nbr_all + (size_t)t * n * MNB,
            out, p_maxit, t, n);
    }
}
